// Encoder_5136780886313
// MI455X (gfx1250) — matrix operand order repaired and checked by reading, not compiled, not run on hardware
//
#include <hip/hip_runtime.h>

// ---------------------------------------------------------------------------
// MI455X (gfx1250) Encoder.
// Heavy GEMMs (KNN Gram, EdgeConv, FFN) use v_wmma_f32_16x16x32_bf16.
//  - 4-wave workgroups: waves cover 4 M-tiles of one 64-wide N-strip; the
//    B panel (4KB / k-chunk) is staged to LDS once per block with async
//    global_load_async_to_lds_b128 + ping-pong double buffering (ASYNCcnt),
//    consumed via ds_load_b128.
//  - Gram/FFN operands pre-converted to bf16 once per layer (pure b128+WMMA).
//  - EdgeConv builds (nbr-ctr, ctr) fragments inline with v_cvt_pk_bf16_f32.
//  - Weights pre-packed into B-fragment order.
// ---------------------------------------------------------------------------

#define USE_ASYNC 1

typedef __attribute__((ext_vector_type(16))) __bf16 v16bf;
typedef __attribute__((ext_vector_type(2)))  __bf16 v2bf;
typedef __attribute__((ext_vector_type(8)))  float  v8f;

union Frag16 {
  v16bf v;
  unsigned short u[16];
  unsigned w[8];
  uint4 q[2];
};

__device__ __forceinline__ unsigned short f2bf(float f) {
  union { float f; unsigned u; } x; x.f = f;
  unsigned r = x.u + 0x7FFFu + ((x.u >> 16) & 1u);   // RNE
  return (unsigned short)(r >> 16);
}

// two fp32 -> packed pair of bf16 (low = a, high = b)
__device__ __forceinline__ unsigned pk2(float a, float b) {
#if __has_builtin(__builtin_amdgcn_cvt_pk_bf16_f32)
  union { v2bf v; unsigned u; } r;
  r.v = __builtin_amdgcn_cvt_pk_bf16_f32(a, b);
  return r.u;
#else
  return (unsigned)f2bf(a) | ((unsigned)f2bf(b) << 16);
#endif
}

// 8 fp32 (two float4) -> bf16 frag positions [base..base+7], base in {0,8}
__device__ __forceinline__ void cvt8(Frag16& f, int base, float4 x, float4 y) {
  const int w0 = base >> 1;
  f.w[w0+0] = pk2(x.x, x.y);
  f.w[w0+1] = pk2(x.z, x.w);
  f.w[w0+2] = pk2(y.x, y.y);
  f.w[w0+3] = pk2(y.z, y.w);
}

#if USE_ASYNC
// CDNA5 async global->LDS copy (16B), tracked by ASYNCcnt.
__device__ __forceinline__ void async_ld_b128(void* lds, const void* g) {
  unsigned off = (unsigned)(uintptr_t)lds;   // low 32 bits = LDS byte offset
  asm volatile("global_load_async_to_lds_b128 %0, %1, off"
               :: "v"(off), "v"(g) : "memory");
}
__device__ __forceinline__ void wait_async() {
  asm volatile("s_wait_asynccnt 0x0" ::: "memory");
}
#endif

#define BN_INV 0.99999500003749993f   // 1/sqrt(1+1e-5)

// ---------------------------------------------------------------------------
// fp32 -> bf16 bulk conversion (4 elements / thread).
// ---------------------------------------------------------------------------
__global__ void conv_bf16v(const float* __restrict__ in,
                           unsigned short* __restrict__ out, int n4) {
  int i = blockIdx.x * blockDim.x + threadIdx.x;
  if (i >= n4) return;
  float4 x = ((const float4*)in)[i];
  uint2 o;
  o.x = pk2(x.x, x.y);
  o.y = pk2(x.z, x.w);
  ((uint2*)out)[i] = o;
}

// ---------------------------------------------------------------------------
// Layer-1 scalar kernels (C=3, negligible FLOPs).
// ---------------------------------------------------------------------------
__global__ void negdist3(const float* __restrict__ X, float* __restrict__ dist, int N) {
  int i = blockIdx.x * 256 + threadIdx.x;        // i over N*N
  if (i >= N * N) return;
  int n = i / N, m = i - n * N;
  const float* Xb = X + (size_t)blockIdx.y * N * 3;
  float dx = Xb[n*3+0] - Xb[m*3+0];
  float dy = Xb[n*3+1] - Xb[m*3+1];
  float dz = Xb[n*3+2] - Xb[m*3+2];
  dist[((size_t)blockIdx.y * N + n) * N + m] = -(dx*dx + dy*dy + dz*dz);
}

__global__ void edgeconv3(const float* __restrict__ X, const int* __restrict__ idx,
                          const float* __restrict__ W,   // (32, 6) row-major
                          const float* __restrict__ g, const float* __restrict__ bia,
                          float* __restrict__ Y, int N) {
  int i = blockIdx.x * 256 + threadIdx.x;        // i over N*32
  if (i >= N * 32) return;
  int n = i >> 5, o = i & 31;
  int b = blockIdx.y;
  const float* Xb = X + (size_t)b * N * 3;
  const int* ib = idx + ((size_t)b * N + n) * 4;
  float cx = Xb[n*3], cy = Xb[n*3+1], cz = Xb[n*3+2];
  const float* w = W + o * 6;
  float wb = w[3]*cx + w[4]*cy + w[5]*cz;
  float gg = g[o] * BN_INV, bb = bia[o];
  float mx = -3e38f;
#pragma unroll
  for (int j = 0; j < 4; ++j) {
    int nb = ib[j];
    float h = w[0]*(Xb[nb*3]-cx) + w[1]*(Xb[nb*3+1]-cy) + w[2]*(Xb[nb*3+2]-cz) + wb;
    h = h * gg + bb;
    h = (h >= 0.f) ? h : 0.2f * h;
    mx = fmaxf(mx, h);
  }
  Y[((size_t)b * N + n) * 32 + o] = mx;
}

// ---------------------------------------------------------------------------
// xx[i] = sum_c X[i,c]^2
// ---------------------------------------------------------------------------
__global__ void row_norm(const float* __restrict__ X, float* __restrict__ xx,
                         int rows, int C) {
  int i = blockIdx.x * blockDim.x + threadIdx.x;
  if (i >= rows) return;
  const float* p = X + (size_t)i * C;
  float s = 0.f;
  for (int c = 0; c < C; ++c) s += p[c] * p[c];
  xx[i] = s;
}

// ---------------------------------------------------------------------------
// Gram / neg-distance via WMMA on bf16 X (C % 32 == 0).
// 4 waves x 16 rows share a 64-col B panel staged in LDS (async, ping-pong).
// grid: (N/64, N/64, B), block 128.
// ---------------------------------------------------------------------------
__global__ void gram_negdist_wmma(const unsigned short* __restrict__ Xbf, // (B,N,C)
                                  const float* __restrict__ xx,
                                  float* __restrict__ dist,
                                  int N, int C) {
  __shared__ unsigned short Bs[2][4][32][16];   // 8KB ping-pong B panel
  const int tid  = threadIdx.x;
  const int wave = tid >> 5, lane = tid & 31;
  const int b  = blockIdx.z;
  const int n0 = blockIdx.x * 64;
  const int m0 = blockIdx.y * 64 + wave * 16;
  const unsigned short* Xb = Xbf + (size_t)b * N * C;
  const int rowA = m0 + (lane & 15);
  const int kA = (lane < 16) ? 0 : 16;
  // staging: wave w stages tile w; lane's fragment source in X
  const int scol = n0 + wave * 16 + (lane & 15);
  const unsigned short* sbase = Xb + (size_t)scol * C + ((lane < 16) ? 0 : 16);
  auto stage = [&](int k0, int buf) {
#if USE_ASYNC
    async_ld_b128(&Bs[buf][wave][lane][0], sbase + k0);
    async_ld_b128(&Bs[buf][wave][lane][8], sbase + k0 + 8);
#else
    uint4* d = (uint4*)&Bs[buf][wave][lane][0];
    d[0] = *(const uint4*)(sbase + k0);
    d[1] = *(const uint4*)(sbase + k0 + 8);
#endif
  };
  v8f acc[4] = {};
  stage(0, 0);
  int buf = 0;
  for (int k0 = 0; k0 < C; k0 += 32) {
#if USE_ASYNC
    wait_async();
#endif
    __syncthreads();
    if (k0 + 32 < C) stage(k0 + 32, buf ^ 1);
    Frag16 a;
    const unsigned short* ap = Xb + (size_t)rowA * C + k0 + kA;
    a.q[0] = *(const uint4*)ap;
    a.q[1] = *(const uint4*)(ap + 8);
    __builtin_prefetch(ap + 32, 0, 1);
#pragma unroll
    for (int t = 0; t < 4; ++t) {
      const uint4* ls = (const uint4*)&Bs[buf][t][lane][0];
      Frag16 bf;
      bf.q[0] = ls[0];
      bf.q[1] = ls[1];
      acc[t] = __builtin_amdgcn_wmma_f32_16x16x32_bf16(false, a.v, false, bf.v,
                                                       (short)0, acc[t], false, false);
    }
    buf ^= 1;
  }
  const float* xxb = xx + (size_t)b * N;
  float* Db = dist + (size_t)b * N * N;
#pragma unroll
  for (int t = 0; t < 4; ++t) {
    const int cc = n0 + t * 16 + (lane & 15);
    const float xc = xxb[cc];
#pragma unroll
    for (int r = 0; r < 8; ++r) {
      int rr = m0 + r + ((lane < 16) ? 0 : 8);
      Db[(size_t)rr * N + cc] = 2.f * acc[t][r] - xxb[rr] - xc;
    }
  }
}

// ---------------------------------------------------------------------------
// Row-wise top-4. grid (N, B), block 32.
// ---------------------------------------------------------------------------
__global__ void topk4(const float* __restrict__ dist, int* __restrict__ idx, int N) {
  __shared__ float sv[128];
  __shared__ int   si[128];
  const int n = blockIdx.x, b = blockIdx.y, lane = threadIdx.x;
  const float* row = dist + ((size_t)b * N + n) * N;
  float bv[4] = {-3e38f, -3e38f, -3e38f, -3e38f};
  int   bi[4] = {0, 0, 0, 0};
  for (int m = lane; m < N; m += 32) {
    float val = row[m];
    if (val > bv[3]) {
      int j = 3;
      while (j > 0 && val > bv[j - 1]) { bv[j] = bv[j-1]; bi[j] = bi[j-1]; --j; }
      bv[j] = val; bi[j] = m;
    }
  }
#pragma unroll
  for (int t = 0; t < 4; ++t) { sv[lane * 4 + t] = bv[t]; si[lane * 4 + t] = bi[t]; }
  __syncthreads();
  if (lane == 0) {
    float rv[4] = {-3e38f, -3e38f, -3e38f, -3e38f};
    int   ri[4] = {0, 0, 0, 0};
    for (int t = 0; t < 128; ++t) {
      float val = sv[t]; int ii = si[t];
      if (val > rv[3] || (val == rv[3] && ii < ri[3])) {
        int j = 3;
        while (j > 0 && (val > rv[j-1] || (val == rv[j-1] && ii < ri[j-1]))) {
          rv[j] = rv[j-1]; ri[j] = ri[j-1]; --j;
        }
        rv[j] = val; ri[j] = ii;
      }
    }
    int* o = idx + ((size_t)b * N + n) * 4;
    for (int t = 0; t < 4; ++t) o[t] = ri[t];
  }
}

// ---------------------------------------------------------------------------
// Pack K x O weights into WMMA B-fragment order (zero-padded past K):
//   P[(((nt*KT)+kt)*32 + lane)*16 + e] = W(k = kt*32 + (lane<16?0:8)+e+(e<8?0:8),
//                                         o = nt*16 + (lane&15))
// TRANS=1: W is (O,K) row-major; TRANS=0: (K,O) row-major.
// ---------------------------------------------------------------------------
template <int TRANS>
__global__ void packB(const float* __restrict__ W, unsigned short* __restrict__ P,
                      int K, int O, int KT) {
  int i = blockIdx.x * blockDim.x + threadIdx.x;
  int total = (O / 16) * KT * 512;
  if (i >= total) return;
  int e = i & 15, lane = (i >> 4) & 31, t = i >> 9;
  int kt = t % KT, nt = t / KT;
  int k = kt * 32 + ((lane < 16) ? 0 : 8) + e + ((e < 8) ? 0 : 8);
  int o = nt * 16 + (lane & 15);
  float v = 0.f;
  if (k < K) v = TRANS ? W[(size_t)o * K + k] : W[(size_t)k * O + o];
  P[i] = f2bf(v);
}

// ---------------------------------------------------------------------------
// Fused EdgeConv (C % 32 == 0): 4 waves x (4 pts x 4 nbrs) x 64 outputs.
// Packed weight panel staged in LDS (async ping-pong); A gathered inline with
// packed bf16 converts. Epilogue: BN + LeakyReLU + max over neighbors.
// grid: (O/64, N/16, B), block 128.
// ---------------------------------------------------------------------------
__global__ void edgeconv_wmma(const float* __restrict__ X,           // (B,N,C)
                              const int* __restrict__ idx,           // (B,N,4)
                              const unsigned short* __restrict__ Bp, // packed (2C,O)
                              const float* __restrict__ g,
                              const float* __restrict__ bia,
                              float* __restrict__ Y,                 // (B,N,O)
                              int N, int C, int O) {
  __shared__ unsigned short Bs[2][4][32][16];
  const int tid  = threadIdx.x;
  const int wave = tid >> 5, lane = tid & 31;
  const int b  = blockIdx.z;
  const int o0 = blockIdx.x * 64;
  const int r0 = blockIdx.y * 64 + wave * 16;     // row = n*4 + j
  const float* Xb = X + (size_t)b * N * C;
  const int* idxb = idx + (size_t)b * N * 4;
  const int K = 2 * C;
  const int KT = K >> 5;
  const int rA = r0 + (lane & 15);
  const int p  = rA >> 2;
  const int nb = idxb[rA];
  const float* ctr = Xb + (size_t)p * C;
  const float* nbr = Xb + (size_t)nb * C;
  const int kA = (lane < 16) ? 0 : 8;
  // staging: wave w stages col-tile w of the packed panel
  const unsigned short* sbase =
      Bp + ((size_t)((o0 >> 4) + wave) * KT) * 512 + lane * 16;
  auto stage = [&](int kt, int buf) {
#if USE_ASYNC
    async_ld_b128(&Bs[buf][wave][lane][0], sbase + (size_t)kt * 512);
    async_ld_b128(&Bs[buf][wave][lane][8], sbase + (size_t)kt * 512 + 8);
#else
    uint4* d = (uint4*)&Bs[buf][wave][lane][0];
    d[0] = *(const uint4*)(sbase + (size_t)kt * 512);
    d[1] = *(const uint4*)(sbase + (size_t)kt * 512 + 8);
#endif
  };
  v8f acc[4] = {};
  stage(0, 0);
  int buf = 0;
  for (int kt = 0; kt < KT; ++kt) {
#if USE_ASYNC
    wait_async();
#endif
    __syncthreads();
    if (kt + 1 < KT) stage(kt + 1, buf ^ 1);
    const int k0 = kt * 32;
    Frag16 a;
    if (k0 < C) {                                  // diff region (uniform branch)
      const float* pn0 = nbr + k0 + kA;
      const float* pc0 = ctr + k0 + kA;
      float4 n0a = *(const float4*)pn0,        n1a = *(const float4*)(pn0 + 4);
      float4 c0a = *(const float4*)pc0,        c1a = *(const float4*)(pc0 + 4);
      float4 n0b = *(const float4*)(pn0 + 16), n1b = *(const float4*)(pn0 + 20);
      float4 c0b = *(const float4*)(pc0 + 16), c1b = *(const float4*)(pc0 + 20);
      float4 d0 = {n0a.x-c0a.x, n0a.y-c0a.y, n0a.z-c0a.z, n0a.w-c0a.w};
      float4 d1 = {n1a.x-c1a.x, n1a.y-c1a.y, n1a.z-c1a.z, n1a.w-c1a.w};
      float4 d2 = {n0b.x-c0b.x, n0b.y-c0b.y, n0b.z-c0b.z, n0b.w-c0b.w};
      float4 d3 = {n1b.x-c1b.x, n1b.y-c1b.y, n1b.z-c1b.z, n1b.w-c1b.w};
      cvt8(a, 0, d0, d1);
      cvt8(a, 8, d2, d3);
    } else {                                       // center region
      const float* pc0 = ctr + (k0 - C) + kA;
      cvt8(a, 0, *(const float4*)pc0,        *(const float4*)(pc0 + 4));
      cvt8(a, 8, *(const float4*)(pc0 + 16), *(const float4*)(pc0 + 20));
    }
#pragma unroll
    for (int t = 0; t < 4; ++t) {
      const uint4* ls = (const uint4*)&Bs[buf][t][lane][0];
      Frag16 bf;
      bf.q[0] = ls[0];
      bf.q[1] = ls[1];
      acc[t] = __builtin_amdgcn_wmma_f32_16x16x32_bf16(false, a.v, false, bf.v,
                                                       (short)0, acc[t], false, false);
    }
    buf ^= 1;
  }
  const int pbase = (r0 >> 2) + ((lane < 16) ? 0 : 2);
  float* Yb = Y + (size_t)b * N * O;
#pragma unroll
  for (int t = 0; t < 4; ++t) {
    const int cB = o0 + t * 16 + (lane & 15);
    const float gg = g[cB] * BN_INV;
    const float bb = bia[cB];
    float h0 = -3e38f, h1 = -3e38f;
#pragma unroll
    for (int r = 0; r < 8; ++r) {
      float h = acc[t][r] * gg + bb;
      h = (h >= 0.f) ? h : 0.2f * h;
      if (r < 4) h0 = fmaxf(h0, h); else h1 = fmaxf(h1, h);
    }
    Yb[(size_t)pbase * O + cB]       = h0;
    Yb[(size_t)(pbase + 1) * O + cB] = h1;
  }
}

// ---------------------------------------------------------------------------
// WMMA GEMM on bf16 A (row-major) x packed B: C = A*B + bias, optional relu,
// fp32 or bf16 output. 4 waves x 16 rows share one 64-col panel (LDS staged).
// grid: (N/64, M/64), block 128.
// ---------------------------------------------------------------------------
template <int RELU, int OUTBF>
__global__ void gemm_wmma(const unsigned short* __restrict__ Ab,
                          const unsigned short* __restrict__ Bp,  // packed
                          const float* __restrict__ bias,
                          void* __restrict__ Out,
                          int M, int N, int K) {
  __shared__ unsigned short Bs[2][4][32][16];
  const int tid  = threadIdx.x;
  const int wave = tid >> 5, lane = tid & 31;
  const int n0 = blockIdx.x * 64;
  const int m0 = blockIdx.y * 64 + wave * 16;
  const int KT = K >> 5;
  const int rowA = m0 + (lane & 15);
  const int kA = (lane < 16) ? 0 : 8;
  const unsigned short* sbase =
      Bp + ((size_t)((n0 >> 4) + wave) * KT) * 512 + lane * 16;
  auto stage = [&](int kt, int buf) {
#if USE_ASYNC
    async_ld_b128(&Bs[buf][wave][lane][0], sbase + (size_t)kt * 512);
    async_ld_b128(&Bs[buf][wave][lane][8], sbase + (size_t)kt * 512 + 8);
#else
    uint4* d = (uint4*)&Bs[buf][wave][lane][0];
    d[0] = *(const uint4*)(sbase + (size_t)kt * 512);
    d[1] = *(const uint4*)(sbase + (size_t)kt * 512 + 8);
#endif
  };
  v8f acc[4] = {};
  stage(0, 0);
  int buf = 0;
  for (int kt = 0; kt < KT; ++kt) {
#if USE_ASYNC
    wait_async();
#endif
    __syncthreads();
    if (kt + 1 < KT) stage(kt + 1, buf ^ 1);
    Frag16 a;
    const unsigned short* ap = Ab + (size_t)rowA * K + kt * 32 + kA;
    a.q[0] = *(const uint4*)ap;
    a.q[1] = *(const uint4*)(ap + 16);
    __builtin_prefetch(ap + 32, 0, 1);
#pragma unroll
    for (int t = 0; t < 4; ++t) {
      const uint4* ls = (const uint4*)&Bs[buf][t][lane][0];
      Frag16 bf;
      bf.q[0] = ls[0];
      bf.q[1] = ls[1];
      acc[t] = __builtin_amdgcn_wmma_f32_16x16x32_bf16(false, a.v, false, bf.v,
                                                       (short)0, acc[t], false, false);
    }
    buf ^= 1;
  }
#pragma unroll
  for (int t = 0; t < 4; ++t) {
    const int colB = n0 + t * 16 + (lane & 15);
    const float bv = bias[colB];
#pragma unroll
    for (int r = 0; r < 8; ++r) {
      int rr = m0 + r + ((lane < 16) ? 0 : 8);
      float v = acc[t][r] + bv;
      if (RELU) v = fmaxf(v, 0.f);
      if (OUTBF) ((unsigned short*)Out)[(size_t)rr * N + colB] = f2bf(v);
      else       ((float*)Out)[(size_t)rr * N + colB] = v;
    }
  }
}

// ---------------------------------------------------------------------------
// q/k/v projections (F -> 1). grid: B*N blocks, 256 threads.
// ---------------------------------------------------------------------------
__global__ void qkv_proj(const float* __restrict__ X,
                         const float* __restrict__ Wq, const float* __restrict__ bq,
                         const float* __restrict__ Wk, const float* __restrict__ bk,
                         const float* __restrict__ Wv, const float* __restrict__ bv,
                         float* __restrict__ q, float* __restrict__ k,
                         float* __restrict__ v, int F) {
  __shared__ float s0[256], s1[256], s2[256];
  const int i = blockIdx.x, t = threadIdx.x;
  const float* xr = X + (size_t)i * F;
  float sq = 0.f, sk = 0.f, sv = 0.f;
  for (int c = t; c < F; c += 256) {
    float xv = xr[c];
    sq += xv * Wq[c]; sk += xv * Wk[c]; sv += xv * Wv[c];
  }
  s0[t] = sq; s1[t] = sk; s2[t] = sv;
  __syncthreads();
  for (int o = 128; o > 0; o >>= 1) {
    if (t < o) { s0[t] += s0[t+o]; s1[t] += s1[t+o]; s2[t] += s2[t+o]; }
    __syncthreads();
  }
  if (t == 0) { q[i] = s0[0] + bq[0]; k[i] = s1[0] + bk[0]; v[i] = s2[0] + bv[0]; }
}

// ---------------------------------------------------------------------------
// h[b,n] = softmax_m(q_n * k_m) . v   (head dim 1, scale 1). grid (N,B), 32.
// ---------------------------------------------------------------------------
__global__ void attn_h(const float* __restrict__ q, const float* __restrict__ k,
                       const float* __restrict__ v, float* __restrict__ h, int N) {
  __shared__ float red[32];
  const int n = blockIdx.x, b = blockIdx.y, lane = threadIdx.x;
  const float* kb = k + (size_t)b * N;
  const float* vb = v + (size_t)b * N;
  const float qn = q[(size_t)b * N + n];
  float mx = -3e38f;
  for (int m = lane; m < N; m += 32) mx = fmaxf(mx, qn * kb[m]);
  red[lane] = mx; __syncthreads();
  for (int o = 16; o > 0; o >>= 1) { if (lane < o) red[lane] = fmaxf(red[lane], red[lane+o]); __syncthreads(); }
  mx = red[0]; __syncthreads();
  float se = 0.f, sv = 0.f;
  for (int m = lane; m < N; m += 32) {
    float e = __expf(qn * kb[m] - mx);
    se += e; sv += e * vb[m];
  }
  red[lane] = se; __syncthreads();
  for (int o = 16; o > 0; o >>= 1) { if (lane < o) red[lane] += red[lane+o]; __syncthreads(); }
  se = red[0]; __syncthreads();
  red[lane] = sv; __syncthreads();
  for (int o = 16; o > 0; o >>= 1) { if (lane < o) red[lane] += red[lane+o]; __syncthreads(); }
  if (lane == 0) h[(size_t)b * N + n] = red[0] / se;
}

// ---------------------------------------------------------------------------
// Residual + LayerNorm.
// MODE 0: s = h[i]*Wo[c] + bo[c] + A[i,c];  MODE 1: s = A[i,c] + Bm[i,c].
// ---------------------------------------------------------------------------
template <int MODE>
__global__ void resid_ln(const float* __restrict__ A, const float* __restrict__ Bm,
                         const float* __restrict__ h,
                         const float* __restrict__ Wo, const float* __restrict__ bo,
                         const float* __restrict__ gamma, const float* __restrict__ beta,
                         float* __restrict__ Y, int F) {
  __shared__ float s1[256], s2[256];
  const int i = blockIdx.x, t = threadIdx.x;
  const float hv = (MODE == 0) ? h[i] : 0.f;
  float sum = 0.f, sq = 0.f;
  for (int c = t; c < F; c += 256) {
    float s = (MODE == 0) ? (hv * Wo[c] + bo[c] + A[(size_t)i * F + c])
                          : (A[(size_t)i * F + c] + Bm[(size_t)i * F + c]);
    sum += s; sq += s * s;
  }
  s1[t] = sum; s2[t] = sq;
  __syncthreads();
  for (int o = 128; o > 0; o >>= 1) {
    if (t < o) { s1[t] += s1[t+o]; s2[t] += s2[t+o]; }
    __syncthreads();
  }
  const float mean = s1[0] / F;
  const float var  = s2[0] / F - mean * mean;
  const float inv  = rsqrtf(var + 1e-5f);
  for (int c = t; c < F; c += 256) {
    float s = (MODE == 0) ? (hv * Wo[c] + bo[c] + A[(size_t)i * F + c])
                          : (A[(size_t)i * F + c] + Bm[(size_t)i * F + c]);
    Y[(size_t)i * F + c] = (s - mean) * inv * gamma[c] + beta[c];
  }
}

// ---------------------------------------------------------------------------
// out[b*ldout + c] = max_n X[b,n,c]. grid (ceil(C/256), B).
// ---------------------------------------------------------------------------
__global__ void colmax(const float* __restrict__ X, float* __restrict__ out,
                       int N, int C, int ldout) {
  int c = blockIdx.x * 256 + threadIdx.x;
  int b = blockIdx.y;
  if (c >= C) return;
  const float* Xb = X + (size_t)b * N * C;
  float m = -3e38f;
  for (int n = 0; n < N; ++n) m = fmaxf(m, Xb[(size_t)n * C + c]);
  out[(size_t)b * ldout + c] = m;
}

__global__ void transpose_bnc(const float* __restrict__ X, float* __restrict__ Y,
                              int N, int C) {
  int i = blockIdx.x * 256 + threadIdx.x;
  if (i >= N * C) return;
  int n = i / C, c = i - n * C;
  int b = blockIdx.y;
  Y[((size_t)b * C + c) * N + n] = X[((size_t)b * N + n) * C + c];
}

__global__ void elemmax(const float* __restrict__ a, const float* __restrict__ b,
                        float* __restrict__ o, int n) {
  int i = blockIdx.x * blockDim.x + threadIdx.x;
  if (i < n) o[i] = fmaxf(a[i], b[i]);
}

// ---------------------------------------------------------------------------
// Host orchestration.
// ---------------------------------------------------------------------------
extern "C" void kernel_launch(void* const* d_in, const int* in_sizes, int n_in,
                              void* d_out, int out_size, void* d_ws, size_t ws_size,
                              hipStream_t stream) {
  (void)in_sizes; (void)n_in; (void)out_size; (void)ws_size;
  const int B = 8, N = 2048, F = 1024, DFF = 2048;
  const int odims[6] = {32, 64, 128, 256, 512, 1024};
  const int cins[6]  = {3, 32, 64, 128, 256, 512};

  char* ws = (char*)d_ws;
  size_t off = 0;
  auto alloc = [&](size_t bytes) -> void* {
    void* p = ws + off;
    off += (bytes + 255) & ~(size_t)255;
    return p;
  };
  float* dist = (float*)alloc((size_t)B * N * N * 4);
  float* act[6];
  for (int l = 0; l < 6; ++l) act[l] = (float*)alloc((size_t)B * N * odims[l] * 4);
  float* xx  = (float*)alloc((size_t)B * N * 4);
  int*   idx = (int*)alloc((size_t)B * N * 4 * 4);
  float* q   = (float*)alloc((size_t)B * N * 4);
  float* k   = (float*)alloc((size_t)B * N * 4);
  float* v   = (float*)alloc((size_t)B * N * 4);
  float* h   = (float*)alloc((size_t)B * N * 4);
  float* xln = (float*)alloc((size_t)B * N * F * 4);
  unsigned short* xbf  = (unsigned short*)alloc((size_t)B * N * 1024 * 2);
  unsigned short* g1   = (unsigned short*)alloc((size_t)B * N * DFF * 2);
  unsigned short* wt   = (unsigned short*)alloc((size_t)2048 * 1024 * 2);
  unsigned short* wf1b = (unsigned short*)alloc((size_t)F * DFF * 2);
  unsigned short* wf2b = (unsigned short*)alloc((size_t)DFF * F * 2);
  float* pool1 = (float*)alloc((size_t)B * 1920 * 4);
  float* pool2 = (float*)alloc((size_t)B * 1920 * 4);

  auto run_extractor = [&](const float* xin, const float* const* P,
                           float* pooled, float* c1dst, float* c2dst) {
    // ---- layer 1 (C=3): scalar path ----
    negdist3<<<dim3((N * N + 255) / 256, B), 256, 0, stream>>>(xin, dist, N);
    topk4<<<dim3(N, B), 32, 0, stream>>>(dist, idx, N);
    edgeconv3<<<dim3((N * 32 + 255) / 256, B), 256, 0, stream>>>(
        xin, idx, P[0], P[1], P[2], act[0], N);
    // ---- layers 2..6: WMMA path ----
    const float* cur = act[0];
    for (int l = 1; l < 6; ++l) {
      const int C = cins[l], O = odims[l], K = 2 * C, KT = K / 32;
      row_norm<<<dim3((B * N + 255) / 256), 256, 0, stream>>>(cur, xx, B * N, C);
      conv_bf16v<<<dim3((B * N * C / 4 + 255) / 256), 256, 0, stream>>>(
          cur, xbf, B * N * C / 4);
      gram_negdist_wmma<<<dim3(N / 64, N / 64, B), 128, 0, stream>>>(xbf, xx, dist, N, C);
      topk4<<<dim3(N, B), 32, 0, stream>>>(dist, idx, N);
      packB<1><<<dim3((K * O + 255) / 256), 256, 0, stream>>>(P[3 * l], wt, K, O, KT);
      edgeconv_wmma<<<dim3(O / 64, N / 16, B), 128, 0, stream>>>(
          cur, idx, wt, P[3 * l + 1], P[3 * l + 2], act[l], N, C, O);
      cur = act[l];
    }
    // ---- attention on act[5] (B,N,1024) ----
    const float* const* A = P + 18;  // Wq bq Wk bk Wv bv Wo bo Wf1 bf1 Wf2 bf2 ln1g ln1b ln2g ln2b
    qkv_proj<<<dim3(B * N), 256, 0, stream>>>(act[5], A[0], A[1], A[2], A[3], A[4], A[5],
                                              q, k, v, F);
    attn_h<<<dim3(N, B), 32, 0, stream>>>(q, k, v, h, N);
    resid_ln<0><<<dim3(B * N), 256, 0, stream>>>(act[5], nullptr, h, A[6], A[7],
                                                 A[12], A[13], xln, F);
    packB<0><<<dim3((F * DFF + 255) / 256), 256, 0, stream>>>(A[8],  wf1b, F, DFF, F / 32);
    packB<0><<<dim3((DFF * F + 255) / 256), 256, 0, stream>>>(A[10], wf2b, DFF, F, DFF / 32);
    conv_bf16v<<<dim3((B * N * F / 4 + 255) / 256), 256, 0, stream>>>(
        xln, xbf, B * N * F / 4);
    gemm_wmma<1, 1><<<dim3(DFF / 64, (B * N) / 64), 128, 0, stream>>>(
        xbf, wf1b, A[9], g1, B * N, DFF, F);
    gemm_wmma<0, 0><<<dim3(F / 64, (B * N) / 64), 128, 0, stream>>>(
        g1, wf2b, A[11], act[5], B * N, F, DFF);
    resid_ln<1><<<dim3(B * N), 256, 0, stream>>>(act[5], xln, nullptr, nullptr, nullptr,
                                                 A[14], A[15], act[5], F);
    // ---- pooling: concat(max x3, x4, x5, x6) ----
    colmax<<<dim3(1, B), 256, 0, stream>>>(act[2], pooled + 0,   N, 128, 1920);
    colmax<<<dim3(1, B), 256, 0, stream>>>(act[3], pooled + 128, N, 256, 1920);
    colmax<<<dim3(2, B), 256, 0, stream>>>(act[4], pooled + 384, N, 512, 1920);
    colmax<<<dim3(4, B), 256, 0, stream>>>(act[5], pooled + 896, N, 1024, 1920);
    // ---- c1 = x1^T, c2 = x2^T ----
    transpose_bnc<<<dim3((N * 32 + 255) / 256, B), 256, 0, stream>>>(act[0], c1dst, N, 32);
    transpose_bnc<<<dim3((N * 64 + 255) / 256, B), 256, 0, stream>>>(act[1], c2dst, N, 64);
  };

  const float* x     = (const float*)d_in[0];
  const float* prior = (const float*)d_in[1];
  const float* P1[34]; for (int i = 0; i < 34; ++i) P1[i] = (const float*)d_in[2 + i];
  const float* P2[34]; for (int i = 0; i < 34; ++i) P2[i] = (const float*)d_in[36 + i];

  float* out = (float*)d_out;
  float* c11 = out + 15360;
  float* c12 = c11 + (size_t)B * 32 * N;
  float* c21 = c12 + (size_t)B * 64 * N;
  float* c22 = c21 + (size_t)B * 32 * N;

  run_extractor(x,     P1, pool1, c11, c12);
  run_extractor(prior, P2, pool2, c21, c22);
  elemmax<<<dim3((B * 1920 + 255) / 256), 256, 0, stream>>>(pool1, pool2, out, B * 1920);
}
